// SomLayer_29454885716109
// MI455X (gfx1250) — matrix operand order repaired and checked by reading, not compiled, not run on hardware
//
#include <hip/hip_runtime.h>

typedef __attribute__((ext_vector_type(16))) _Float16 v16h;
typedef __attribute__((ext_vector_type(8)))  float    v8f;

constexpr int SOM_D   = 256;   // feature dim
constexpr int SOM_K   = 1024;  // codebook entries
constexpr int MBLOCK  = 128;   // rows per workgroup (8 waves x 16)
constexpr int KC      = 64;    // codes per LDS chunk
constexpr int NKC     = SOM_K / KC;       // 16
constexpr int DPAD    = 272;   // padded LDS row stride in halves (544B = 17*32B)

#if defined(__AMDGCN__) && __has_builtin(__builtin_amdgcn_global_load_async_to_lds_b128)
#define SOM_ASYNC 1
typedef int v4i_vs __attribute__((vector_size(16)));
typedef __attribute__((address_space(1))) v4i_vs g_v4i;   // global int4
typedef __attribute__((address_space(3))) v4i_vs l_v4i;   // LDS int4
#if __has_builtin(__builtin_amdgcn_s_wait_asynccnt)
#define SOM_WAIT_ASYNC() __builtin_amdgcn_s_wait_asynccnt(0)
#else
#define SOM_WAIT_ASYNC() asm volatile("s_wait_asynccnt 0" ::: "memory")
#endif
#endif

// ---------------------------------------------------------------------------
// Prep: codebook f32 -> f16 (row-major, contiguous) + per-code squared norms.
// One wave per code; grid = SOM_K/8 blocks of 256 threads.
// ---------------------------------------------------------------------------
__global__ void som_prep(const float* __restrict__ cb,
                         _Float16* __restrict__ hB,
                         float* __restrict__ norms) {
  const int lane = threadIdx.x & 31;
  const int wave = threadIdx.x >> 5;
  const int k    = blockIdx.x * 8 + wave;
  const float* row = cb + (size_t)k * SOM_D;
  float s = 0.0f;
#pragma unroll
  for (int j = 0; j < 8; ++j) {
    float v = row[j * 32 + lane];
    hB[(size_t)k * SOM_D + j * 32 + lane] = (_Float16)v;
    s += v * v;
  }
#pragma unroll
  for (int off = 16; off >= 1; off >>= 1) s += __shfl_xor(s, off, 32);
  if (lane == 0) norms[k] = s;
}

// ---------------------------------------------------------------------------
// Main: WMMA-tiled distance matmul + argmin + straight-through gather.
// 256 threads (8 waves); each wave owns a 16-row A tile resident in VGPRs.
// Codebook chunks double-buffered in LDS via async-to-LDS DMA.
// ---------------------------------------------------------------------------
__global__ void __launch_bounds__(256)
som_main(const float* __restrict__ x,
         const float* __restrict__ cb,
         const _Float16* __restrict__ hB,
         const float* __restrict__ norms,
         float* __restrict__ out) {
  __shared__ alignas(32) _Float16 sB[2][KC * DPAD];   // 2 x ~34 KB padded f16 tiles
  __shared__ int sBmu[MBLOCK];

  const int tid  = threadIdx.x;
  const int lane = tid & 31;
  const int wave = tid >> 5;
  const int hgrp = lane >> 4;           // 0 = lanes 0-15, 1 = lanes 16-31
  const int m    = lane & 15;
  const int row0 = blockIdx.x * MBLOCK;
  const int myRow = row0 + wave * 16 + m;

  // ---- stage one 64x256 f16 codebook chunk into LDS buffer `buf`
  auto stage = [&](int kc, int buf) {
#pragma unroll
    for (int it = 0; it < (KC * SOM_D) / (8 * 256); ++it) {   // 8 iters of 16B
      int idx = it * 256 + tid;
      int r   = idx >> 5;                 // code row within chunk
      int u   = idx & 31;                 // 16B unit within row
      const _Float16* src = hB + (size_t)(kc * KC + r) * SOM_D + u * 8;
      _Float16* dst = &sB[buf][r * DPAD + u * 8];
#ifdef SOM_ASYNC
      __builtin_amdgcn_global_load_async_to_lds_b128((g_v4i*)src, (l_v4i*)dst, 0, 0);
#else
      *(float4*)dst = *(const float4*)src;
#endif
    }
  };

  // ---- A tile: 16 rows x 256 (f32 -> f16), contiguous k = 16*hgrp + s per lane, kept in VGPRs
  stage(0, 0);   // kick off first chunk before the (long) A load+convert
  v16h a[8];
  {
    const float* xr = x + (size_t)myRow * SOM_D;
    const int kb = hgrp * 16;
#pragma unroll
    for (int c = 0; c < 8; ++c) {
      const float* p0 = xr + c * 32 + kb;        // K = c*32 + kb + [0..8)
      const float* p1 = p0 + 8;                  // K = c*32 + kb + 8 + [0..8)
      union { v16h v; _Float16 h[16]; } u;
#pragma unroll
      for (int j = 0; j < 8; ++j) u.h[j] = (_Float16)p0[j];
#pragma unroll
      for (int j = 0; j < 8; ++j) u.h[8 + j] = (_Float16)p1[j];
      a[c] = u.v;
    }
  }

  float bestv[8];
  int   besti[8];
#pragma unroll
  for (int r = 0; r < 8; ++r) { bestv[r] = __builtin_inff(); besti[r] = 0; }

  for (int kc = 0; kc < NKC; ++kc) {
    const int buf = kc & 1;
#ifdef SOM_ASYNC
    SOM_WAIT_ASYNC();          // this wave's chunk-kc DMA done
#endif
    __syncthreads();           // all waves' DMA done + prev compute on other buf done

    if (kc + 1 < NKC) stage(kc + 1, buf ^ 1);   // overlap next DMA with compute

#pragma unroll
    for (int t = 0; t < KC / 16; ++t) {   // 4 code tiles of 16
      v8f acc = {};
#pragma unroll
      for (int c = 0; c < 8; ++c) {
        // B fragment: column n = m, K = c*32 + hgrp*16 + [0..16), 32B aligned
        const v16h b = *(const v16h*)&sB[buf][(t * 16 + m) * DPAD + c * 32 + hgrp * 16];
        acc = __builtin_amdgcn_wmma_f32_16x16x32_f16(
            false, a[c], false, b, (short)0, acc, false, false);
      }
      const int   nidx = kc * KC + t * 16 + m;
      const float nrm  = norms[nidx];
#pragma unroll
      for (int r = 0; r < 8; ++r) {
        float dv = __builtin_fmaf(-2.0f, acc[r], nrm);   // ||e||^2 - 2 x.e
        bool better = dv < bestv[r];                     // strict: keeps lowest n on ties
        bestv[r] = better ? dv : bestv[r];
        besti[r] = better ? nidx : besti[r];
      }
    }
  }

  // ---- argmin across the 16 lanes holding one matrix row (xor stays in half)
#pragma unroll
  for (int r = 0; r < 8; ++r) {
    float v = bestv[r];
    int   i = besti[r];
#pragma unroll
    for (int off = 8; off >= 1; off >>= 1) {
      float vo = __shfl_xor(v, off, 32);
      int   io = __shfl_xor(i, off, 32);
      bool take = (vo < v) || (vo == v && io < i);
      v = take ? vo : v;
      i = take ? io : i;
    }
    if (m == 0) sBmu[wave * 16 + hgrp * 8 + r] = i;   // M = r + 8*hgrp
  }
  __syncthreads();

  // ---- gather + straight-through rounding: out = x + (q - x)
  const float4* cb4  = (const float4*)cb;
  const float4* x4   = (const float4*)x;
  float4*       out4 = (float4*)out;
#pragma unroll 4
  for (int it = 0; it < (MBLOCK * (SOM_D / 4)) / 256; ++it) {   // 32 iters
    int idx = it * 256 + tid;
    int r   = idx >> 6;          // local row 0..127
    int d4  = idx & 63;          // float4 within row
    int code = sBmu[r];
    float4 q  = cb4[(size_t)code * (SOM_D / 4) + d4];
    float4 xv = x4[(size_t)(row0 + r) * (SOM_D / 4) + d4];
    float4 o;
    o.x = xv.x + (q.x - xv.x);
    o.y = xv.y + (q.y - xv.y);
    o.z = xv.z + (q.z - xv.z);
    o.w = xv.w + (q.w - xv.w);
    out4[(size_t)(row0 + r) * (SOM_D / 4) + d4] = o;
  }
}

// ---------------------------------------------------------------------------
extern "C" void kernel_launch(void* const* d_in, const int* in_sizes, int n_in,
                              void* d_out, int out_size, void* d_ws, size_t ws_size,
                              hipStream_t stream) {
  const float* x  = (const float*)d_in[0];   // [8,4096,256] f32
  const float* cb = (const float*)d_in[1];   // [1024,256]   f32
  float* out = (float*)d_out;

  _Float16* hB    = (_Float16*)d_ws;                                   // 512 KB
  float*    norms = (float*)((char*)d_ws +
                             (size_t)SOM_K * SOM_D * sizeof(_Float16)); // 4 KB

  som_prep<<<SOM_K / 8, 256, 0, stream>>>(cb, hB, norms);

  const int N = in_sizes[0] / SOM_D;         // 32768 rows
  som_main<<<N / MBLOCK, 256, 0, stream>>>(x, cb, hB, norms, out);
}
